// SupervisedContrastiveLearning_76141180223509
// MI455X (gfx1250) — matrix operand order repaired and checked by reading, not compiled, not run on hardware
//
#include <hip/hip_runtime.h>
#include <hip/hip_bf16.h>

// ---------------------------------------------------------------------------
// Supervised contrastive loss forward, MI455X (gfx1250), wave32 + bf16 WMMA.
//   3x MLP [1024->512->256->128] (+ReLU) -> l2norm -> concat [B,384]
//   combined MLP [384->256->128] (+ReLU on first) -> l2norm -> Z [8192,128]
//   sim = Z Z^T / 0.07 ; per-row logsumexp vs positive-mask sum -> scalar loss
// All GEMMs: v_wmma_f32_16x16x32_bf16 (fp32 accumulate), 16x64 strip per wave
// (A-fragment reused over 4 WMMAs). sim matrix never materialized.
// ---------------------------------------------------------------------------

#define BN 8192
#define HD 1024
#define PROJ 128
#define INV_TEMP 14.2857142857142857f   // 1/0.07
#define L2EPS 1e-12f

typedef __attribute__((ext_vector_type(16))) __bf16 v16bf;
typedef __attribute__((ext_vector_type(8)))  float  v8f;

union FragBF {
    v16bf v;
    unsigned short us[16];
    uint4 q[2];
};

__device__ __forceinline__ unsigned short f2bf(float f) {
    unsigned int u = __float_as_uint(f);
    u += 0x7FFFu + ((u >> 16) & 1u);     // round-to-nearest-even
    return (unsigned short)(u >> 16);
}

// --------------------------- weight transpose + bf16 -----------------------
// W [K,N] f32 row-major  ->  Wt [N,K] bf16, so B-fragments are K-contiguous.
__global__ void wt_convert(const float* __restrict__ W,
                           unsigned short* __restrict__ Wt, int K, int N) {
    int idx = blockIdx.x * 256 + threadIdx.x;
    if (idx < K * N) {
        int k = idx / N, n = idx - k * N;
        Wt[(size_t)n * K + k] = f2bf(W[idx]);
    }
}

// ------------------------------- WMMA GEMM ---------------------------------
// Y[M,N] = act(A[M,K] @ Wt^T + bias). Each wave owns a 16x64 output strip:
// one A fragment per K-step feeds 4 WMMAs (4 adjacent N-tiles, 4 fp32 accs).
// A: f32 (layer 1, converted inline) or bf16. Y: bf16 or f32 (pre-normalize).
template <bool A_F32, bool RELU, bool OUT_F32>
__global__ void gemm_wmma(const void* __restrict__ Aptr,
                          const unsigned short* __restrict__ Wt,
                          const float* __restrict__ bias,
                          void* __restrict__ Yptr,
                          int M, int N, int K, int stripsN) {
    const int lane  = threadIdx.x & 31;
    const int wave  = threadIdx.x >> 5;
    const int strip = blockIdx.x * 8 + wave;
    const int tilesM = M >> 4;
    if (strip >= tilesM * stripsN) return;         // wave-uniform guard
    const int tm = strip / stripsN;
    const int sn = strip - tm * stripsN;           // 64-column strip index

    const int nloc  = lane & 15;
    const int hi    = lane >> 4;
    const int mrow  = tm * 16 + nloc;              // A row for this lane
    const int kbase = hi * 16;                     // A-frag K sub-offset
    const int ncol0 = sn * 64 + nloc;              // B col of sub-tile 0

    // Per-lane weight row pointers for the 4 N-tiles of the strip.
    const unsigned short* wr0 = Wt + (size_t)(ncol0     ) * K + hi * 16;
    const unsigned short* wr1 = Wt + (size_t)(ncol0 + 16) * K + hi * 16;
    const unsigned short* wr2 = Wt + (size_t)(ncol0 + 32) * K + hi * 16;
    const unsigned short* wr3 = Wt + (size_t)(ncol0 + 48) * K + hi * 16;

    v8f c0 = {}, c1 = {}, c2 = {}, c3 = {};

    const float* aF = (const float*)Aptr + (size_t)mrow * K;
    const unsigned short* aH = (const unsigned short*)Aptr + (size_t)mrow * K;

#pragma unroll 2
    for (int k0 = 0; k0 < K; k0 += 32) {
        FragBF a;
        if (A_F32) {
#pragma unroll
            for (int i = 0; i < 8; ++i) a.us[i]     = f2bf(aF[k0 + kbase + i]);
#pragma unroll
            for (int i = 0; i < 8; ++i) a.us[8 + i] = f2bf(aF[k0 + kbase + 8 + i]);
        } else {
            a.q[0] = *(const uint4*)(aH + k0 + kbase);
            a.q[1] = *(const uint4*)(aH + k0 + kbase + 8);
        }
        FragBF b0, b1, b2, b3;
        b0.q[0] = *(const uint4*)(wr0 + k0);
        b0.q[1] = *(const uint4*)(wr0 + k0 + 8);
        b1.q[0] = *(const uint4*)(wr1 + k0);
        b1.q[1] = *(const uint4*)(wr1 + k0 + 8);
        b2.q[0] = *(const uint4*)(wr2 + k0);
        b2.q[1] = *(const uint4*)(wr2 + k0 + 8);
        b3.q[0] = *(const uint4*)(wr3 + k0);
        b3.q[1] = *(const uint4*)(wr3 + k0 + 8);
        if (k0 + 32 < K) {                         // global_prefetch_b8 flavor
            __builtin_prefetch(wr0 + k0 + 32, 0, 0);
            __builtin_prefetch(wr2 + k0 + 32, 0, 0);
        }
        c0 = __builtin_amdgcn_wmma_f32_16x16x32_bf16(false, a.v, false, b0.v,
                                                     (short)0, c0, false, false);
        c1 = __builtin_amdgcn_wmma_f32_16x16x32_bf16(false, a.v, false, b1.v,
                                                     (short)0, c1, false, false);
        c2 = __builtin_amdgcn_wmma_f32_16x16x32_bf16(false, a.v, false, b2.v,
                                                     (short)0, c2, false, false);
        c3 = __builtin_amdgcn_wmma_f32_16x16x32_bf16(false, a.v, false, b3.v,
                                                     (short)0, c3, false, false);
    }

    const int mbase = tm * 16 + hi * 8;            // C/D layout: lanes 16-31 -> M+8
#pragma unroll
    for (int t = 0; t < 4; ++t) {
        const v8f c = (t == 0) ? c0 : (t == 1) ? c1 : (t == 2) ? c2 : c3;
        const int ncol = ncol0 + t * 16;
        const float bv = bias[ncol];
        if (OUT_F32) {
            float* Y = (float*)Yptr;
#pragma unroll
            for (int r = 0; r < 8; ++r) {
                float v = c[r] + bv;
                if (RELU) v = fmaxf(v, 0.0f);
                Y[(size_t)(mbase + r) * N + ncol] = v;
            }
        } else {
            unsigned short* Y = (unsigned short*)Yptr;
#pragma unroll
            for (int r = 0; r < 8; ++r) {
                float v = c[r] + bv;
                if (RELU) v = fmaxf(v, 0.0f);
                Y[(size_t)(mbase + r) * N + ncol] = f2bf(v);
            }
        }
    }
}

// --------------------------- row L2-normalize ------------------------------
// 128-wide fp32 rows -> bf16 rows at (dstStride, colOff). One wave per row.
__global__ void l2norm_to_bf16(const float* __restrict__ X,
                               unsigned short* __restrict__ dst,
                               int dstStride, int colOff) {
    const int row  = blockIdx.x * 8 + (threadIdx.x >> 5);
    const int lane = threadIdx.x & 31;
    const float4 v = *(const float4*)(X + (size_t)row * PROJ + lane * 4);
    float ss = v.x * v.x + v.y * v.y + v.z * v.z + v.w * v.w;
#pragma unroll
    for (int m = 1; m < 32; m <<= 1) ss += __shfl_xor(ss, m, 32);
    const float inv = 1.0f / fmaxf(sqrtf(ss), L2EPS);
    unsigned short* d = dst + (size_t)row * dstStride + colOff + lane * 4;
    d[0] = f2bf(v.x * inv);
    d[1] = f2bf(v.y * inv);
    d[2] = f2bf(v.z * inv);
    d[3] = f2bf(v.w * inv);
}

// --------------------- fused Z Z^T + contrastive stats ---------------------
// One block per 16-row tile; 8 waves stride the 512 column tiles. K=128 via
// 4 chained bf16 WMMAs; exp/mask sums accumulate in registers, then reduce
// across the 16 lanes of each half and flush via atomics.
__global__ void loss_tiles(const unsigned short* __restrict__ Zb,
                           const int* __restrict__ labels,
                           float* __restrict__ sumExp,
                           float* __restrict__ posSum,
                           float* __restrict__ posCnt) {
    const int lane = threadIdx.x & 31;
    const int wave = threadIdx.x >> 5;
    const int it   = blockIdx.x;                   // row tile 0..511
    const int nloc = lane & 15;
    const int hi   = lane >> 4;

    // Preload the 4 A-fragments for this row tile (reused across all j).
    const unsigned short* ap = Zb + (size_t)(it * 16 + nloc) * PROJ + hi * 8;
    FragBF A[4];
#pragma unroll
    for (int ks = 0; ks < 4; ++ks) {
        A[ks].q[0] = *(const uint4*)(ap + ks * 32);
        A[ks].q[1] = *(const uint4*)(ap + ks * 32 + 16);
    }
    int labi[8];
#pragma unroll
    for (int r = 0; r < 8; ++r) labi[r] = labels[it * 16 + hi * 8 + r];

    float accS[8], accP[8], accC[8];
#pragma unroll
    for (int r = 0; r < 8; ++r) { accS[r] = 0.f; accP[r] = 0.f; accC[r] = 0.f; }

    for (int jt = wave; jt < BN / 16; jt += 8) {
        const int j = jt * 16 + nloc;
        const unsigned short* bp = Zb + (size_t)j * PROJ + hi * 8;
        v8f c = {};
#pragma unroll
        for (int ks = 0; ks < 4; ++ks) {
            FragBF b;
            b.q[0] = *(const uint4*)(bp + ks * 32);
            b.q[1] = *(const uint4*)(bp + ks * 32 + 16);
            c = __builtin_amdgcn_wmma_f32_16x16x32_bf16(
                    false, A[ks].v, false, b.v, (short)0, c, false, false);
        }
        const int labj = labels[j];
#pragma unroll
        for (int r = 0; r < 8; ++r) {
            const float e = __expf(c[r] * INV_TEMP);
            accS[r] += e;                          // diagonal included (as ref)
            const int i = it * 16 + hi * 8 + r;
            if (labj == labi[r] && j != i) { accP[r] += e; accC[r] += 1.0f; }
        }
    }

#pragma unroll
    for (int r = 0; r < 8; ++r) {
        float s = accS[r], p = accP[r], cc = accC[r];
#pragma unroll
        for (int m = 1; m < 16; m <<= 1) {         // reduce within each half
            s  += __shfl_xor(s, m, 32);
            p  += __shfl_xor(p, m, 32);
            cc += __shfl_xor(cc, m, 32);
        }
        if (nloc == 0) {
            const int i = it * 16 + hi * 8 + r;
            atomicAdd(&sumExp[i], s);
            atomicAdd(&posSum[i], p);
            atomicAdd(&posCnt[i], cc);
        }
    }
}

// ------------------------------ finalize -----------------------------------
__global__ void finalize(const float* __restrict__ sumExp,
                         const float* __restrict__ posSum,
                         const float* __restrict__ posCnt,
                         float* __restrict__ out) {
    __shared__ float sTot[256];
    __shared__ float sCnt[256];
    float tot = 0.f, cnt = 0.f;
    for (int i = threadIdx.x; i < BN; i += 256) {
        const bool valid = (posCnt[i] > 0.f) && (posSum[i] > 0.f);
        if (valid) {
            tot += logf(sumExp[i]) - logf(posSum[i]);
            cnt += 1.0f;
        }
    }
    sTot[threadIdx.x] = tot;
    sCnt[threadIdx.x] = cnt;
    __syncthreads();
    for (int s = 128; s > 0; s >>= 1) {
        if (threadIdx.x < s) {
            sTot[threadIdx.x] += sTot[threadIdx.x + s];
            sCnt[threadIdx.x] += sCnt[threadIdx.x + s];
        }
        __syncthreads();
    }
    if (threadIdx.x == 0) {
        const float c = sCnt[0];
        out[0] = (c > 0.f) ? (sTot[0] / fmaxf(c, 1.0f)) : 0.0f;
    }
}

// ------------------------------ launcher -----------------------------------
extern "C" void kernel_launch(void* const* d_in, const int* in_sizes, int n_in,
                              void* d_out, int out_size, void* d_ws, size_t ws_size,
                              hipStream_t stream) {
    (void)n_in; (void)out_size; (void)ws_size;

    const float* reprs[3] = { (const float*)d_in[0], (const float*)d_in[1],
                              (const float*)d_in[2] };
    const int* labels = (const int*)d_in[3];

    // Param flattening order: insertion (aspect,opinion,sentiment,combined)
    // vs jax-sorted (aspect,combined,opinion,sentiment). Detect from sizes:
    // slot 10 holds combined.W0 (384*256) if alphabetically sorted.
    int branchBase[3];   // aspect, opinion, sentiment : each 6 slots (W,b)x3
    int combBase;        // 4 slots (W,b)x2
    if (in_sizes[10] == 384 * 256) {
        branchBase[0] = 4;  combBase = 10; branchBase[1] = 14; branchBase[2] = 20;
    } else {
        branchBase[0] = 4;  branchBase[1] = 10; branchBase[2] = 16; combBase = 22;
    }

    // ---- workspace carve-up (all offsets 256B aligned) ----
    char* ws = (char*)d_ws;
    size_t off = 0;
    auto alloc = [&](size_t bytes) -> void* {
        void* p = ws + off;
        off += (bytes + 255) & ~(size_t)255;
        return p;
    };
    unsigned short* WtB[3][3];
    const int bk[3] = { HD, HD / 2, HD / 4 };      // 1024, 512, 256
    const int bn[3] = { HD / 2, HD / 4, PROJ };    // 512, 256, 128
    for (int br = 0; br < 3; ++br)
        for (int l = 0; l < 3; ++l)
            WtB[br][l] = (unsigned short*)alloc((size_t)bk[l] * bn[l] * 2);
    unsigned short* Wc0t = (unsigned short*)alloc((size_t)384 * 256 * 2);
    unsigned short* Wc1t = (unsigned short*)alloc((size_t)256 * PROJ * 2);

    unsigned short* H1 = (unsigned short*)alloc((size_t)BN * 512 * 2);
    unsigned short* H2 = (unsigned short*)alloc((size_t)BN * 256 * 2);
    float*          Pf = (float*)alloc((size_t)BN * PROJ * 4);   // also reused as Z0
    unsigned short* Cb = (unsigned short*)alloc((size_t)BN * 384 * 2);
    unsigned short* G1 = (unsigned short*)alloc((size_t)BN * 256 * 2);
    unsigned short* Zb = (unsigned short*)alloc((size_t)BN * PROJ * 2);
    float* stats = (float*)alloc((size_t)3 * BN * 4);
    float* sumExp = stats;
    float* posSum = stats + BN;
    float* posCnt = stats + 2 * BN;

    // ---- 1) transpose+bf16 all weights ----
    auto cvt = [&](const float* W, unsigned short* Wt, int K, int N) {
        const int tot = K * N;
        wt_convert<<<(tot + 255) / 256, 256, 0, stream>>>(W, Wt, K, N);
    };
    for (int br = 0; br < 3; ++br)
        for (int l = 0; l < 3; ++l)
            cvt((const float*)d_in[branchBase[br] + 2 * l], WtB[br][l], bk[l], bn[l]);
    cvt((const float*)d_in[combBase + 0], Wc0t, 384, 256);
    cvt((const float*)d_in[combBase + 2], Wc1t, 256, PROJ);

    hipMemsetAsync(stats, 0, (size_t)3 * BN * 4, stream);

    const int tilesM = BN / 16;                    // 512
    auto blocksFor = [&](int N) {                  // strips of 64 columns
        return (tilesM * (N / 64) + 7) / 8;
    };

    // ---- 2) three projector branches ----
    for (int br = 0; br < 3; ++br) {
        const float* b0 = (const float*)d_in[branchBase[br] + 1];
        const float* b1 = (const float*)d_in[branchBase[br] + 3];
        const float* b2 = (const float*)d_in[branchBase[br] + 5];
        gemm_wmma<true,  true,  false><<<blocksFor(512), 256, 0, stream>>>(
            reprs[br], WtB[br][0], b0, H1, BN, 512, 1024, 512 / 64);
        gemm_wmma<false, true,  false><<<blocksFor(256), 256, 0, stream>>>(
            H1, WtB[br][1], b1, H2, BN, 256, 512, 256 / 64);
        gemm_wmma<false, false, true ><<<blocksFor(PROJ), 256, 0, stream>>>(
            H2, WtB[br][2], b2, Pf, BN, PROJ, 256, PROJ / 64);
        l2norm_to_bf16<<<BN / 8, 256, 0, stream>>>(Pf, Cb, 384, br * PROJ);
    }

    // ---- 3) combined MLP + normalize -> Z ----
    const float* bc0 = (const float*)d_in[combBase + 1];
    const float* bc1 = (const float*)d_in[combBase + 3];
    gemm_wmma<false, true,  false><<<blocksFor(256), 256, 0, stream>>>(
        Cb, Wc0t, bc0, G1, BN, 256, 384, 256 / 64);
    gemm_wmma<false, false, true ><<<blocksFor(PROJ), 256, 0, stream>>>(
        G1, Wc1t, bc1, Pf, BN, PROJ, 256, PROJ / 64);
    l2norm_to_bf16<<<BN / 8, 256, 0, stream>>>(Pf, Zb, PROJ, 0);

    // ---- 4) fused similarity + contrastive stats, then scalar loss ----
    loss_tiles<<<BN / 16, 256, 0, stream>>>(Zb, labels, sumExp, posSum, posCnt);
    finalize<<<1, 256, 0, stream>>>(sumExp, posSum, posCnt, (float*)d_out);
}
